// MPNN_47038481826180
// MI455X (gfx1250) — compile-verified, matrix operand order checked by reading, not run on hardware
//
#include <hip/hip_runtime.h>
#include <hip/hip_bf16.h>

// ---------------------------------------------------------------------------
// MPNN (GRU message passing) for MI455X / gfx1250.
// All GEMMs on v_wmma_f32_16x16x32_bf16 (bf16 in, f32 accumulate).
// Weights pre-swizzled into WMMA fragment order -> B frags are 2x b128 loads.
// A frags come from LDS as 2x ds_load_b128. SELU is branch-free.
// ---------------------------------------------------------------------------

#define BB 16      // batch
#define NN 1024    // nodes
#define DD 128     // feature dim
#define EE 16384   // edges
#define AA 64      // actions
#define KPAD 288   // 257 padded up to multiple of 32 (zero rows)

typedef __attribute__((ext_vector_type(16))) __bf16 v16bf;
typedef __attribute__((ext_vector_type(8)))  __bf16 v8bf;
typedef __attribute__((ext_vector_type(8)))  float  v8f;

__device__ __forceinline__ v8f wmma_bf16(v16bf a, v16bf b, v8f c) {
  // 8 args: (neg_a, A, neg_b, B, c_mod, C, reuse_a, reuse_b)
  return __builtin_amdgcn_wmma_f32_16x16x32_bf16(false, a, false, b,
                                                 (short)0, c, false, false);
}

// A-fragment (16x32 bf16, MxK) from a row-major bf16 LDS tile.
// Lane l (0..15): row m=l, K = 0..7 | 16..23; lanes 16..31: K = 8..15 | 24..31.
// Each half is 8 contiguous bf16 (16B) -> two ds_load_b128 per lane.
// Requires: base 16B-aligned, rowstride a multiple of 8 elements.
__device__ __forceinline__ v16bf load_frag_A(const __bf16* base, int rowstride) {
  const int lane = threadIdx.x & 31;
  const __bf16* p = base + (lane & 15) * rowstride + ((lane >> 4) << 3);
  v8bf lo = *(const v8bf*)(p);
  v8bf hi = *(const v8bf*)(p + 16);
  return __builtin_shufflevector(lo, hi, 0, 1, 2, 3, 4, 5, 6, 7,
                                         8, 9, 10, 11, 12, 13, 14, 15);
}

// B-fragment from pre-swizzled weights: frag-major [frag][lane][16 elems],
// 512 elements (1 KB) per fragment -> two global_load_b128 per lane.
__device__ __forceinline__ v16bf load_frag_Bs(const __bf16* frag_base) {
  const int lane = threadIdx.x & 31;
  const __bf16* p = frag_base + lane * 16;
  v8bf lo = *(const v8bf*)(p);
  v8bf hi = *(const v8bf*)(p + 8);
  return __builtin_shufflevector(lo, hi, 0, 1, 2, 3, 4, 5, 6, 7,
                                         8, 9, 10, 11, 12, 13, 14, 15);
}

// Branch-free SELU: for x>0 the exp term is exp(0)-1 == 0; for x<=0 max()==0.
__device__ __forceinline__ float selu_f(float x) {
  const float sc = 1.0507009873554805f, al = 1.6732632423543772f;
  return sc * fmaxf(x, 0.f) + (sc * al) * (__expf(fminf(x, 0.f)) - 1.f);
}
__device__ __forceinline__ float sigmoid_f(float x) {
  return 1.f / (1.f + __expf(-x));
}

// ---------------------------------------------------------------------------
// Weight prep: f32 -> bf16, swizzled into WMMA B-fragment order.
// Fragment f covers K-block kb (32 rows) x N-tile nt (16 cols); within a
// fragment, lane l & element e map to K = kb*32 + (e&7) + 16*(e>>3) + 8*(l>>4),
// N = nt*16 + (l&15).  W_ih/W_hh are transposed on the fly (gi = agg @ W^T).
// ---------------------------------------------------------------------------
__global__ void MPNN_prep_kernel(const float* __restrict__ Wmsg,
                                 const float* __restrict__ Wih,
                                 const float* __restrict__ Whh,
                                 const float* __restrict__ Wr1,
                                 const float* __restrict__ Wr2,
                                 const float* __restrict__ Wpol,
                                 __bf16* __restrict__ Wmsg_s,
                                 __bf16* __restrict__ Wih_s,
                                 __bf16* __restrict__ Whh_s,
                                 __bf16* __restrict__ Wr1_s,
                                 __bf16* __restrict__ Wr2_s,
                                 __bf16* __restrict__ Wpol_s) {
  const int i = blockIdx.x * blockDim.x + threadIdx.x;
  const int lane = (i >> 4) & 31;
  const int e    = i & 15;
  const int kin  = (e & 7) + ((e >> 3) << 4) + ((lane >> 4) << 3);
  const int nin  = lane & 15;

  if (i < 9 * 8 * 512) {               // Wmsg: 9 K-blocks x 8 N-tiles (K padded)
    const int f = i >> 9, kb = f >> 3, nt = f & 7;
    const int K = kb * 32 + kin, N = nt * 16 + nin;
    Wmsg_s[i] = (__bf16)(K < (2 * DD + 1) ? Wmsg[K * DD + N] : 0.0f);
  }
  if (i < 4 * 24 * 512) {              // Wih/Whh: 4 K-blocks x 24 col-tiles
    const int f = i >> 9, kb = f / 24, nt = f % 24;
    const int K = kb * 32 + kin, J = nt * 16 + nin;
    Wih_s[i] = (__bf16)Wih[J * DD + K];   // transpose: B(k,j) = W[j,k]
    Whh_s[i] = (__bf16)Whh[J * DD + K];
  }
  if (i < 4 * 8 * 512) {               // Wr1/Wr2: 4 x 8
    const int f = i >> 9, kb = f >> 3, nt = f & 7;
    const int K = kb * 32 + kin, N = nt * 16 + nin;
    Wr1_s[i] = (__bf16)Wr1[K * DD + N];
    Wr2_s[i] = (__bf16)Wr2[K * DD + N];
  }
  if (i < 4 * 4 * 512) {               // Wpol: 4 x 4
    const int f = i >> 9, kb = f >> 2, nt = f & 3;
    const int K = kb * 32 + kin, N = nt * 16 + nin;
    Wpol_s[i] = (__bf16)Wpol[K * AA + N];
  }
}

// ---------------------------------------------------------------------------
// Edge-feature gather: ef[b,e] = edge[b, src[e], tgt[e]]  (loop-invariant)
// ---------------------------------------------------------------------------
__global__ void MPNN_ef_kernel(const float* __restrict__ edge,
                               const int* __restrict__ src,
                               const int* __restrict__ tgt,
                               float* __restrict__ ef) {
  const int i = blockIdx.x * blockDim.x + threadIdx.x;
  if (i >= BB * EE) return;
  const int b = i >> 14;               // EE = 2^14
  const int e = i & (EE - 1);
  ef[i] = edge[(size_t)b * NN * NN + (size_t)src[e] * NN + tgt[e]];
}

// ---------------------------------------------------------------------------
// Message kernel: msg = selu([h[src]|h[tgt]|ef|0] @ Wmsg + b_msg), scatter-add
// Grid: B * E/16 blocks, 128 threads (4 waves; wave w owns output cols 32w..)
// ---------------------------------------------------------------------------
__global__ __launch_bounds__(128) void MPNN_msg_kernel(
    const float* __restrict__ h, const float* __restrict__ ef,
    const int* __restrict__ src, const int* __restrict__ tgt,
    const __bf16* __restrict__ Wmsg_s, const float* __restrict__ bmsg,
    float* __restrict__ agg) {
  __shared__ __align__(16) __bf16 As[16][KPAD + 8];  // row stride 296 (16B mult)
  __shared__ int s_src[16], s_tgt[16];

  const int tid = threadIdx.x;
  const int b  = blockIdx.x >> 10;     // E/16 = 1024 tiles
  const int e0 = (blockIdx.x & 1023) << 4;

  if (tid < 16) {
    s_src[tid] = src[e0 + tid];
    s_tgt[tid] = tgt[e0 + tid];
  }
  __syncthreads();

  // Cooperative staging: 8 threads per edge row, 16 K each; vector bf16 stores.
  const int m  = tid >> 3;
  const int k0 = (tid & 7) << 4;
  const float* hs = h + ((size_t)b * NN + s_src[m]) * DD;
  const float* ht = h + ((size_t)b * NN + s_tgt[m]) * DD;
#pragma unroll
  for (int g = 0; g < 2; ++g) {
    v8bf ts, tt;
#pragma unroll
    for (int k = 0; k < 8; ++k) {
      ts[k] = (__bf16)hs[k0 + g * 8 + k];
      tt[k] = (__bf16)ht[k0 + g * 8 + k];
    }
    *(v8bf*)&As[m][k0 + g * 8]      = ts;
    *(v8bf*)&As[m][DD + k0 + g * 8] = tt;
  }
  if ((tid & 7) == 0) {
    As[m][2 * DD] = (__bf16)ef[(size_t)b * EE + e0 + m];
#pragma unroll
    for (int k = 2 * DD + 1; k < KPAD; ++k) As[m][k] = (__bf16)0.0f;
  }
  __syncthreads();

  const int wave = tid >> 5;
  const int n0   = wave * 32;
  v8f c0 = {}, c1 = {};
#pragma unroll
  for (int kb = 0; kb < KPAD / 32; ++kb) {
    v16bf a  = load_frag_A(&As[0][kb * 32], KPAD + 8);
    v16bf b0 = load_frag_Bs(Wmsg_s + ((size_t)(kb * 8 + 2 * wave)     << 9));
    v16bf b1 = load_frag_Bs(Wmsg_s + ((size_t)(kb * 8 + 2 * wave + 1) << 9));
    c0 = wmma_bf16(a, b0, c0);
    c1 = wmma_bf16(a, b1, c1);
  }

  // SELU + scatter-add to agg[b, tgt, :] with HW f32 atomics.
  const int lane  = tid & 31;
  const int col   = lane & 15;
  const int rbase = (lane >> 4) << 3;
  const float bm0 = bmsg[n0 + col];
  const float bm1 = bmsg[n0 + 16 + col];
#pragma unroll
  for (int i = 0; i < 8; ++i) {
    const int row = rbase + i;
    float* aggp = agg + ((size_t)b * NN + s_tgt[row]) * DD;
    unsafeAtomicAdd(&aggp[n0 + col],      selu_f(c0[i] + bm0));
    unsafeAtomicAdd(&aggp[n0 + 16 + col], selu_f(c1[i] + bm1));
  }
}

// ---------------------------------------------------------------------------
// GRU kernel: gi = agg@Wih^T + b_ih, gh = h@Whh^T + b_hh, gated update (in place)
// Grid: B * N/16 blocks, 128 threads. Each wave owns 6 of 24 column tiles.
// ---------------------------------------------------------------------------
__global__ __launch_bounds__(128) void MPNN_gru_kernel(
    float* __restrict__ h, const float* __restrict__ agg,
    const __bf16* __restrict__ Wih_s, const __bf16* __restrict__ Whh_s,
    const float* __restrict__ bih, const float* __restrict__ bhh) {
  __shared__ __align__(16) __bf16 hA[16][DD + 8];
  __shared__ __align__(16) __bf16 gA[16][DD + 8];
  __shared__ float gi[16][3 * DD + 8];
  __shared__ float gh[16][3 * DD + 8];

  const int tid = threadIdx.x;
  const int b   = blockIdx.x >> 6;     // N/16 = 64 tiles
  const int nd0 = (blockIdx.x & 63) << 4;

  const int m  = tid >> 3;
  const int k0 = (tid & 7) << 4;
  const float* hp = h   + ((size_t)b * NN + nd0 + m) * DD;
  const float* gp = agg + ((size_t)b * NN + nd0 + m) * DD;
#pragma unroll
  for (int g = 0; g < 2; ++g) {
    v8bf th, tg;
#pragma unroll
    for (int k = 0; k < 8; ++k) {
      th[k] = (__bf16)hp[k0 + g * 8 + k];
      tg[k] = (__bf16)gp[k0 + g * 8 + k];
    }
    *(v8bf*)&hA[m][k0 + g * 8] = th;
    *(v8bf*)&gA[m][k0 + g * 8] = tg;
  }
  __syncthreads();

  const int wave  = tid >> 5;
  const int lane  = tid & 31;
  const int col16 = lane & 15;
  const int rbase = (lane >> 4) << 3;
  for (int t = 0; t < 6; ++t) {
    const int nt = wave * 6 + t;
    v8f ci = {}, ch = {};
#pragma unroll
    for (int kb = 0; kb < DD / 32; ++kb) {
      v16bf ag = load_frag_A(&gA[0][kb * 32], DD + 8);
      v16bf ah = load_frag_A(&hA[0][kb * 32], DD + 8);
      v16bf bi = load_frag_Bs(Wih_s + ((size_t)(kb * 24 + nt) << 9));
      v16bf bh = load_frag_Bs(Whh_s + ((size_t)(kb * 24 + nt) << 9));
      ci = wmma_bf16(ag, bi, ci);
      ch = wmma_bf16(ah, bh, ch);
    }
    const int col = nt * 16 + col16;
    const float bi_c = bih[col], bh_c = bhh[col];
#pragma unroll
    for (int i = 0; i < 8; ++i) {
      gi[rbase + i][col] = ci[i] + bi_c;
      gh[rbase + i][col] = ch[i] + bh_c;
    }
  }
  __syncthreads();

  // Elementwise gates; each thread owns feature column j = tid for 16 nodes.
  const int j = tid;
  float* hw = h + ((size_t)b * NN + nd0) * DD;
  for (int mm = 0; mm < 16; ++mm) {
    const float r  = sigmoid_f(gi[mm][j]          + gh[mm][j]);
    const float z  = sigmoid_f(gi[mm][j + DD]     + gh[mm][j + DD]);
    const float nv = tanhf(    gi[mm][j + 2 * DD] + r * gh[mm][j + 2 * DD]);
    const float ho = hw[mm * DD + j];
    hw[mm * DD + j] = (1.f - z) * nv + z * ho;
  }
}

// ---------------------------------------------------------------------------
// Pool: pooled[b,d] = sum_n h[b,n,d].   Grid: B blocks, 128 threads.
// ---------------------------------------------------------------------------
__global__ __launch_bounds__(128) void MPNN_pool_kernel(const float* __restrict__ h,
                                                        float* __restrict__ pooled) {
  const int b = blockIdx.x, d = threadIdx.x;
  const float* hp = h + (size_t)b * NN * DD + d;
  float s = 0.f;
  for (int n = 0; n < NN; ++n) s += hp[(size_t)n * DD];
  pooled[b * DD + d] = s;
}

// ---------------------------------------------------------------------------
// Readout: selu(selu(pooled@Wr1+b1)@Wr2+b2)@Wpol+bp.  One wave (M=16=B rows).
// ---------------------------------------------------------------------------
__global__ __launch_bounds__(32) void MPNN_readout_kernel(
    const float* __restrict__ pooled,
    const __bf16* __restrict__ Wr1_s, const float* __restrict__ br1,
    const __bf16* __restrict__ Wr2_s, const float* __restrict__ br2,
    const __bf16* __restrict__ Wpol_s, const float* __restrict__ bpol,
    float* __restrict__ out) {
  __shared__ __align__(16) __bf16 As[16][DD + 8];
  __shared__ float Ts[16][DD + 8];

  const int tid   = threadIdx.x;
  const int col16 = tid & 15;
  const int rbase = (tid >> 4) << 3;

  for (int idx = tid; idx < 16 * DD; idx += 32)
    As[idx >> 7][idx & 127] = (__bf16)pooled[idx];
  __syncthreads();

  // Layer 1: Wr1 (128x128), 8 column tiles
  for (int nt = 0; nt < 8; ++nt) {
    v8f c = {};
#pragma unroll
    for (int kb = 0; kb < DD / 32; ++kb) {
      v16bf a = load_frag_A(&As[0][kb * 32], DD + 8);
      v16bf b = load_frag_Bs(Wr1_s + ((size_t)(kb * 8 + nt) << 9));
      c = wmma_bf16(a, b, c);
    }
    const float bc = br1[nt * 16 + col16];
#pragma unroll
    for (int i = 0; i < 8; ++i) Ts[rbase + i][nt * 16 + col16] = selu_f(c[i] + bc);
  }
  __syncthreads();
  for (int idx = tid; idx < 16 * DD; idx += 32)
    As[idx >> 7][idx & 127] = (__bf16)Ts[idx >> 7][idx & 127];
  __syncthreads();

  // Layer 2: Wr2 (128x128)
  for (int nt = 0; nt < 8; ++nt) {
    v8f c = {};
#pragma unroll
    for (int kb = 0; kb < DD / 32; ++kb) {
      v16bf a = load_frag_A(&As[0][kb * 32], DD + 8);
      v16bf b = load_frag_Bs(Wr2_s + ((size_t)(kb * 8 + nt) << 9));
      c = wmma_bf16(a, b, c);
    }
    const float bc = br2[nt * 16 + col16];
#pragma unroll
    for (int i = 0; i < 8; ++i) Ts[rbase + i][nt * 16 + col16] = selu_f(c[i] + bc);
  }
  __syncthreads();
  for (int idx = tid; idx < 16 * DD; idx += 32)
    As[idx >> 7][idx & 127] = (__bf16)Ts[idx >> 7][idx & 127];
  __syncthreads();

  // Policy head: Wpol (128x64), write (16,64) f32 output
  for (int nt = 0; nt < 4; ++nt) {
    v8f c = {};
#pragma unroll
    for (int kb = 0; kb < DD / 32; ++kb) {
      v16bf a = load_frag_A(&As[0][kb * 32], DD + 8);
      v16bf b = load_frag_Bs(Wpol_s + ((size_t)(kb * 4 + nt) << 9));
      c = wmma_bf16(a, b, c);
    }
    const float bc = bpol[nt * 16 + col16];
#pragma unroll
    for (int i = 0; i < 8; ++i)
      out[(size_t)(rbase + i) * AA + nt * 16 + col16] = c[i] + bc;
  }
}

// ---------------------------------------------------------------------------
// Host-side launch sequence (graph-capture safe: kernels + async memset/memcpy)
// ---------------------------------------------------------------------------
extern "C" void kernel_launch(void* const* d_in, const int* in_sizes, int n_in,
                              void* d_out, int out_size, void* d_ws, size_t ws_size,
                              hipStream_t stream) {
  (void)in_sizes; (void)n_in; (void)out_size; (void)ws_size;

  const float* node_f = (const float*)d_in[0];
  const float* edge_f = (const float*)d_in[1];
  const int*   srcIdx = (const int*)d_in[2];
  const int*   tgtIdx = (const int*)d_in[3];
  const float* W_msg  = (const float*)d_in[4];
  const float* b_msg  = (const float*)d_in[5];
  const float* W_ih   = (const float*)d_in[6];
  const float* W_hh   = (const float*)d_in[7];
  const float* b_ih   = (const float*)d_in[8];
  const float* b_hh   = (const float*)d_in[9];
  const float* W_r1   = (const float*)d_in[10];
  const float* b_r1   = (const float*)d_in[11];
  const float* W_r2   = (const float*)d_in[12];
  const float* b_r2   = (const float*)d_in[13];
  const float* W_pol  = (const float*)d_in[14];
  const float* b_pol  = (const float*)d_in[15];
  float* out = (float*)d_out;

  // Workspace layout (bytes)
  char* ws = (char*)d_ws;
  const size_t SZ_H   = (size_t)BB * NN * DD * sizeof(float);   // 8 MB
  const size_t SZ_AGG = SZ_H;                                   // 8 MB
  const size_t SZ_EF  = (size_t)BB * EE * sizeof(float);        // 1 MB
  const size_t SZ_PL  = (size_t)BB * DD * sizeof(float);
  float* h_ws    = (float*)(ws);
  float* agg_ws  = (float*)(ws + SZ_H);
  float* ef_ws   = (float*)(ws + SZ_H + SZ_AGG);
  float* pool_ws = (float*)(ws + SZ_H + SZ_AGG + SZ_EF);
  char*  wb      = ws + SZ_H + SZ_AGG + SZ_EF + SZ_PL;
  __bf16* Wmsg_s = (__bf16*)(wb);  wb += (size_t)(9 * 8)  * 512 * 2;
  __bf16* Wih_s  = (__bf16*)(wb);  wb += (size_t)(4 * 24) * 512 * 2;
  __bf16* Whh_s  = (__bf16*)(wb);  wb += (size_t)(4 * 24) * 512 * 2;
  __bf16* Wr1_s  = (__bf16*)(wb);  wb += (size_t)(4 * 8)  * 512 * 2;
  __bf16* Wr2_s  = (__bf16*)(wb);  wb += (size_t)(4 * 8)  * 512 * 2;
  __bf16* Wpol_s = (__bf16*)(wb);

  // 1) weights -> bf16, swizzled into fragment order
  MPNN_prep_kernel<<<(4 * 24 * 512 + 255) / 256, 256, 0, stream>>>(
      W_msg, W_ih, W_hh, W_r1, W_r2, W_pol,
      Wmsg_s, Wih_s, Whh_s, Wr1_s, Wr2_s, Wpol_s);

  // 2) gather edge features once
  MPNN_ef_kernel<<<(BB * EE + 255) / 256, 256, 0, stream>>>(edge_f, srcIdx, tgtIdx, ef_ws);

  // 3) h = node_features
  hipMemcpyAsync(h_ws, node_f, SZ_H, hipMemcpyDeviceToDevice, stream);

  // 4) three message-passing iterations
  for (int it = 0; it < 3; ++it) {
    hipMemsetAsync(agg_ws, 0, SZ_AGG, stream);
    MPNN_msg_kernel<<<BB * (EE / 16), 128, 0, stream>>>(
        h_ws, ef_ws, srcIdx, tgtIdx, Wmsg_s, b_msg, agg_ws);
    MPNN_gru_kernel<<<BB * (NN / 16), 128, 0, stream>>>(
        h_ws, agg_ws, Wih_s, Whh_s, b_ih, b_hh);
  }

  // 5) pooled readout
  MPNN_pool_kernel<<<BB, DD, 0, stream>>>(h_ws, pool_ws);
  MPNN_readout_kernel<<<1, 32, 0, stream>>>(
      pool_ws, Wr1_s, b_r1, Wr2_s, b_r2, Wpol_s, b_pol, out);
}
